// NodeModel_824633721180
// MI455X (gfx1250) — matrix operand order repaired and checked by reading, not compiled, not run on hardware
//
#include <hip/hip_runtime.h>

#define HIDDEN 128

typedef __attribute__((ext_vector_type(16))) __bf16 v16bf;
typedef __attribute__((ext_vector_type(8)))  __bf16 v8bf;
typedef __attribute__((ext_vector_type(8)))  float  v8f;

// ---------------------------------------------------------------------------
// Native non-returning fp32 global atomic add (GLOBAL_ATOMIC_ADD_F32, no TH:RET)
// ---------------------------------------------------------------------------
__device__ __forceinline__ void atomic_add_f32_nr(float* p, float v) {
  asm volatile("global_atomic_add_f32 %0, %1, off" : : "v"(p), "v"(v) : "memory");
}

// ---------------------------------------------------------------------------
// Zero an f32 buffer (count in float4 units)
// ---------------------------------------------------------------------------
__global__ void zero_f32(float4* __restrict__ p, long n4) {
  long i = (long)blockIdx.x * blockDim.x + threadIdx.x;
  long stride = (long)gridDim.x * blockDim.x;
  float4 z; z.x = z.y = z.z = z.w = 0.0f;
  for (; i < n4; i += stride) p[i] = z;
}

// ---------------------------------------------------------------------------
// Pack weight matrix W [K x 128] (f32, row-major) into WMMA bf16 B-fragments.
// Fragment tile (kt, nt): lane L, element j  ->  K = kt*32 + (L/16)*8 + j + (j>=8 ? 8 : 0),
//                                               N = nt*16 + (L%16)
// (matches 16-bit B 32x16 layout: lanes 0-15 K=0..7,16..23, lanes 16-31 K=8..15,24..31,
//  two bf16 per VGPR)
// ---------------------------------------------------------------------------
__global__ void pack_w(const float* __restrict__ W, __bf16* __restrict__ dst,
                       int ktiles) {
  int slot = blockIdx.x * blockDim.x + threadIdx.x;   // (kt*8+nt)*32 + lane
  int nslots = ktiles * 8 * 32;
  if (slot >= nslots) return;
  int lane = slot & 31;
  int tile = slot >> 5;
  int nt = tile & 7;
  int kt = tile >> 3;
  int n  = nt * 16 + (lane & 15);
  int kb = kt * 32 + (lane >> 4) * 8;
#pragma unroll
  for (int j = 0; j < 16; ++j)
    dst[(long)slot * 16 + j] = (__bf16)W[(long)(kb + j + (j >= 8 ? 8 : 0)) * HIDDEN + n];
}

// ---------------------------------------------------------------------------
// Scatter-add edge messages into node accumulator: one wave per edge,
// lane owns 4 consecutive features (float4 read, 4 native f32 atomics).
// ---------------------------------------------------------------------------
__global__ __launch_bounds__(256)
void scatter_edges(const float* __restrict__ edge_attr,
                   const int* __restrict__ col,
                   float* __restrict__ agg,
                   int n_edges) {
  int wave = (int)((blockIdx.x * blockDim.x + threadIdx.x) >> 5);
  int lane = threadIdx.x & 31;
  if (wave >= n_edges) return;
  int c = col[wave];
  const float4 v =
      *reinterpret_cast<const float4*>(edge_attr + (long)wave * HIDDEN + lane * 4);
  float* dst = agg + (long)c * HIDDEN + lane * 4;
  atomic_add_f32_nr(dst + 0, v.x);
  atomic_add_f32_nr(dst + 1, v.y);
  atomic_add_f32_nr(dst + 2, v.z);
  atomic_add_f32_nr(dst + 3, v.w);
}

// ---------------------------------------------------------------------------
// Fused MLP:  out = relu([x | agg] @ W1 + b1) @ W2 + b2
// Block = 256 threads = 8 waves; wave w computes rows [rowBase, rowBase+16)
// across all 128 output columns (8 n-tiles) with v_wmma_f32_16x16x32_bf16.
// ---------------------------------------------------------------------------
__global__ __launch_bounds__(256)
void fused_mlp(const float* __restrict__ x, const float* __restrict__ agg,
               const __bf16* __restrict__ w1p, const float* __restrict__ b1,
               const __bf16* __restrict__ w2p, const float* __restrict__ b2,
               float* __restrict__ out, int n_nodes) {
  __shared__ __bf16 hlds[128 * HIDDEN];   // 32 KB: block's hidden activations

  const int tid  = threadIdx.x;
  const int wv   = tid >> 5;
  const int lane = tid & 31;
  const int m    = lane & 15;      // row (A) or col (B/C) within 16
  const int half = lane >> 4;      // K-half selector
  const int rowBase = blockIdx.x * 128 + wv * 16;

  // ---- GEMM1: C init = b1 broadcast (C element (r,lane) has N = nt*16+m) ----
  v8f acc[8];
#pragma unroll
  for (int nt = 0; nt < 8; ++nt) {
    float bb = b1[nt * 16 + m];
#pragma unroll
    for (int j = 0; j < 8; ++j) acc[nt][j] = bb;
  }

  // A row for this lane (clamped for the ragged tail; stores are predicated)
  int rowA  = rowBase + m;
  int rowAc = rowA < n_nodes ? rowA : (n_nodes - 1);
  const float* xrow = x   + (long)rowAc * HIDDEN;
  const float* arow = agg + (long)rowAc * HIDDEN;

#pragma unroll
  for (int kt = 0; kt < 8; ++kt) {
    // A fragment (16-bit A 16x32): lane half picks K+0/K+8; chunks at +0,+16
    int kc0 = kt * 32 + half * 8;
    int kc1 = kc0 + 16;
    const float* s0 = (kc0 < HIDDEN) ? (xrow + kc0) : (arow + (kc0 - HIDDEN));
    const float* s1 = (kc1 < HIDDEN) ? (xrow + kc1) : (arow + (kc1 - HIDDEN));
    float4 c00 = *reinterpret_cast<const float4*>(s0);
    float4 c01 = *reinterpret_cast<const float4*>(s0 + 4);
    float4 c10 = *reinterpret_cast<const float4*>(s1);
    float4 c11 = *reinterpret_cast<const float4*>(s1 + 4);
    v16bf a;
    a[0]  = (__bf16)c00.x; a[1]  = (__bf16)c00.y; a[2]  = (__bf16)c00.z; a[3]  = (__bf16)c00.w;
    a[4]  = (__bf16)c01.x; a[5]  = (__bf16)c01.y; a[6]  = (__bf16)c01.z; a[7]  = (__bf16)c01.w;
    a[8]  = (__bf16)c10.x; a[9]  = (__bf16)c10.y; a[10] = (__bf16)c10.z; a[11] = (__bf16)c10.w;
    a[12] = (__bf16)c11.x; a[13] = (__bf16)c11.y; a[14] = (__bf16)c11.z; a[15] = (__bf16)c11.w;

#pragma unroll
    for (int nt = 0; nt < 8; ++nt) {
      v16bf b = *reinterpret_cast<const v16bf*>(
          w1p + ((long)(kt * 8 + nt) * 32 + lane) * 16);
      acc[nt] = __builtin_amdgcn_wmma_f32_16x16x32_bf16(
          false, a, false, b, (short)0, acc[nt], false, false);
    }
  }

  // ---- ReLU -> bf16 -> LDS (C layout -> row-major; LDS does the transpose) --
#pragma unroll
  for (int nt = 0; nt < 8; ++nt) {
#pragma unroll
    for (int r = 0; r < 8; ++r) {
      float hv = acc[nt][r];
      hv = hv > 0.0f ? hv : 0.0f;
      hlds[(wv * 16 + r + 8 * half) * HIDDEN + nt * 16 + m] = (__bf16)hv;
    }
  }
  __syncthreads();

  // ---- GEMM2: C init = b2 broadcast ----
  v8f acc2[8];
#pragma unroll
  for (int nt = 0; nt < 8; ++nt) {
    float bb = b2[nt * 16 + m];
#pragma unroll
    for (int j = 0; j < 8; ++j) acc2[nt][j] = bb;
  }

  const __bf16* rp = &hlds[(wv * 16 + m) * HIDDEN];
#pragma unroll
  for (int kt = 0; kt < 4; ++kt) {
    int kc0 = kt * 32 + half * 8;
    v8bf lo = *reinterpret_cast<const v8bf*>(rp + kc0);
    v8bf hi = *reinterpret_cast<const v8bf*>(rp + kc0 + 16);
    v16bf a;
#pragma unroll
    for (int j = 0; j < 8; ++j) { a[j] = lo[j]; a[8 + j] = hi[j]; }

#pragma unroll
    for (int nt = 0; nt < 8; ++nt) {
      v16bf b = *reinterpret_cast<const v16bf*>(
          w2p + ((long)(kt * 8 + nt) * 32 + lane) * 16);
      acc2[nt] = __builtin_amdgcn_wmma_f32_16x16x32_bf16(
          false, a, false, b, (short)0, acc2[nt], false, false);
    }
  }

  // ---- Store (C layout: M = r + 8*half, N = nt*16 + m) ----
#pragma unroll
  for (int nt = 0; nt < 8; ++nt) {
#pragma unroll
    for (int r = 0; r < 8; ++r) {
      int row = rowBase + r + 8 * half;
      if (row < n_nodes) out[(long)row * HIDDEN + nt * 16 + m] = acc2[nt][r];
    }
  }
}

// ---------------------------------------------------------------------------
// Host launcher
// inputs: x, edge_index, edge_attr, u, batch, W1, b1, W2, b2
// ---------------------------------------------------------------------------
extern "C" void kernel_launch(void* const* d_in, const int* in_sizes, int n_in,
                              void* d_out, int out_size, void* d_ws, size_t ws_size,
                              hipStream_t stream) {
  const float* x          = (const float*)d_in[0];
  const int*   edge_index = (const int*)d_in[1];
  const float* edge_attr  = (const float*)d_in[2];
  const float* W1         = (const float*)d_in[5];
  const float* b1         = (const float*)d_in[6];
  const float* W2         = (const float*)d_in[7];
  const float* b2         = (const float*)d_in[8];
  float* out = (float*)d_out;

  const int n_nodes = in_sizes[0] / HIDDEN;
  const int n_edges = in_sizes[1] / 2;
  const int* col = edge_index + n_edges;   // edge_index[1], destination nodes

  // workspace layout: agg f32 [n_nodes*128] | packed W1 bf16 | packed W2 bf16
  char* ws = (char*)d_ws;
  size_t aggBytes = ((size_t)n_nodes * HIDDEN * sizeof(float) + 255) & ~(size_t)255;
  float*  agg = (float*)ws;
  __bf16* w1p = (__bf16*)(ws + aggBytes);          // 8 kt * 8 nt * 512 = 64 KB
  __bf16* w2p = w1p + (size_t)8 * 8 * 32 * 16;     // 4 kt * 8 nt * 512 = 32 KB

  long n4 = (long)n_nodes * HIDDEN / 4;
  zero_f32<<<1024, 256, 0, stream>>>((float4*)agg, n4);

  pack_w<<<(8 * 8 * 32 + 255) / 256, 256, 0, stream>>>(W1, w1p, 8);
  pack_w<<<(4 * 8 * 32 + 255) / 256, 256, 0, stream>>>(W2, w2p, 4);

  int edgeBlocks = (n_edges + 7) / 8;              // 8 waves/block, wave per edge
  scatter_edges<<<edgeBlocks, 256, 0, stream>>>(edge_attr, col, agg, n_edges);

  int nodeBlocks = (n_nodes + 127) / 128;          // 128 rows per block
  fused_mlp<<<nodeBlocks, 256, 0, stream>>>(x, agg, w1p, b1, w2p, b2, out, n_nodes);
}
